// CrfRnnLayer_56667798504225
// MI455X (gfx1250) — matrix operand order repaired and checked by reading, not compiled, not run on hardware
//
#include <hip/hip_runtime.h>
#include <hip/hip_bf16.h>

typedef __attribute__((ext_vector_type(16))) __bf16 v16bf;
typedef __attribute__((ext_vector_type(8)))  float  v8f;

#define HH 96
#define WW 96
#define NN (HH * WW)   // 9216
#define CC 8
#define NITER 5

// ---- workspace layout (bytes) ----
static constexpr size_t OFF_KC  = 0;                               // NN*NN bf16  (~162 MB)
static constexpr size_t OFF_FB  = (size_t)NN * NN * 2;             // NN*8 f32 bilateral feats
static constexpr size_t OFF_UF  = OFF_FB  + (size_t)NN * 8 * 4;    // NN*8 f32 unary [N,C]
static constexpr size_t OFF_Q   = OFF_UF  + (size_t)NN * 8 * 4;    // NN*8 f32 current q
static constexpr size_t OFF_SMT = OFF_Q   + (size_t)NN * 8 * 4;    // 16*NN bf16 softmax^T (padded)
static constexpr size_t OFF_NS  = OFF_SMT + (size_t)16 * NN * 2;   // NN f32 spatial norm
static constexpr size_t OFF_NB  = OFF_NS  + (size_t)NN * 4;        // NN f32 bilateral norm

// -------------------------------------------------------------------------
// 1) Per-pixel prep: bilateral features, unary transpose, q init, smT pad rows
// -------------------------------------------------------------------------
__global__ void crf_prep_kernel(const float* __restrict__ unary,
                                const float* __restrict__ rgb,
                                float* __restrict__ fb,
                                float* __restrict__ uf,
                                float* __restrict__ q,
                                __bf16* __restrict__ smT) {
    int i = blockIdx.x * blockDim.x + threadIdx.x;
    if (i >= NN) return;
    float y = (float)(i / WW), x = (float)(i % WW);
    float r = rgb[i], g = rgb[NN + i], b = rgb[2 * NN + i];
    float* f = fb + (size_t)i * 8;
    f[0] = y * (1.0f / 160.0f);
    f[1] = x * (1.0f / 160.0f);
    f[2] = r * (1.0f / 3.0f);
    f[3] = g * (1.0f / 3.0f);
    f[4] = b * (1.0f / 3.0f);
    f[5] = 0.0f; f[6] = 0.0f; f[7] = 0.0f;
#pragma unroll
    for (int c = 0; c < CC; ++c) {
        float u = unary[(size_t)c * NN + i];
        uf[(size_t)i * 8 + c] = u;
        q [(size_t)i * 8 + c] = u;
    }
    // zero the padded channel rows (8..15) of smT once
#pragma unroll
    for (int c = CC; c < 16; ++c) smT[(size_t)c * NN + i] = (__bf16)0.0f;
}

// -------------------------------------------------------------------------
// 2) Row sums of Ks and Kb (filter response to ones). One block per row i.
// -------------------------------------------------------------------------
__global__ __launch_bounds__(256) void crf_norm_kernel(const float* __restrict__ fb,
                                                       float* __restrict__ ns,
                                                       float* __restrict__ nb) {
    __shared__ float red_s[256];
    __shared__ float red_b[256];
    int i = blockIdx.x, tid = threadIdx.x;
    float yi = (float)(i / WW) * (1.0f / 3.0f);
    float xi = (float)(i % WW) * (1.0f / 3.0f);
    const float4* fi = (const float4*)(fb + (size_t)i * 8);
    float4 a0 = fi[0], a1 = fi[1];
    float ss = 0.0f, sb = 0.0f;
    for (int j = tid; j < NN; j += 256) {
        float dy = yi - (float)(j / WW) * (1.0f / 3.0f);
        float dx = xi - (float)(j % WW) * (1.0f / 3.0f);
        float d2s = dy * dy + dx * dx;
        const float4* fj = (const float4*)(fb + (size_t)j * 8);
        float4 b0 = fj[0], b1 = fj[1];
        float d0 = a0.x - b0.x, d1 = a0.y - b0.y, d2 = a0.z - b0.z;
        float d3 = a0.w - b0.w, d4 = a1.x - b1.x;
        float d2b = d0 * d0 + d1 * d1 + d2 * d2 + d3 * d3 + d4 * d4;
        ss += __expf(-0.5f * d2s);
        sb += __expf(-0.5f * d2b);
    }
    red_s[tid] = ss; red_b[tid] = sb;
    __syncthreads();
    for (int s = 128; s > 0; s >>= 1) {
        if (tid < s) { red_s[tid] += red_s[tid + s]; red_b[tid] += red_b[tid + s]; }
        __syncthreads();
    }
    if (tid == 0) { ns[i] = red_s[0]; nb[i] = red_b[0]; }
}

// -------------------------------------------------------------------------
// 3) Build combined, row-normalized, weighted kernel: Kc (bf16, fits in L2)
// -------------------------------------------------------------------------
__global__ __launch_bounds__(256) void crf_kc_kernel(const float* __restrict__ fb,
                                                     const float* __restrict__ ns,
                                                     const float* __restrict__ nb,
                                                     const float* __restrict__ wsp,
                                                     const float* __restrict__ wbi,
                                                     __bf16* __restrict__ Kc) {
    int i = blockIdx.x, tid = threadIdx.x;
    float cs = wsp[0] / ns[i];
    float cb = wbi[0] / nb[i];
    float yi = (float)(i / WW) * (1.0f / 3.0f);
    float xi = (float)(i % WW) * (1.0f / 3.0f);
    const float4* fi = (const float4*)(fb + (size_t)i * 8);
    float4 a0 = fi[0], a1 = fi[1];
    __bf16* row = Kc + (size_t)i * NN;
    for (int j = tid; j < NN; j += 256) {
        float dy = yi - (float)(j / WW) * (1.0f / 3.0f);
        float dx = xi - (float)(j % WW) * (1.0f / 3.0f);
        float d2s = dy * dy + dx * dx;
        const float4* fj = (const float4*)(fb + (size_t)j * 8);
        float4 b0 = fj[0], b1 = fj[1];
        float d0 = a0.x - b0.x, d1 = a0.y - b0.y, d2 = a0.z - b0.z;
        float d3 = a0.w - b0.w, d4 = a1.x - b1.x;
        float d2b = d0 * d0 + d1 * d1 + d2 * d2 + d3 * d3 + d4 * d4;
        float kc = cs * __expf(-0.5f * d2s) + cb * __expf(-0.5f * d2b);
        row[j] = (__bf16)kc;
    }
}

// -------------------------------------------------------------------------
// 4) Row softmax of q -> transposed bf16 buffer smT [16 x N] (rows 8..15 = 0)
// -------------------------------------------------------------------------
__global__ void crf_softmax_kernel(const float* __restrict__ q,
                                   __bf16* __restrict__ smT) {
    int i = blockIdx.x * blockDim.x + threadIdx.x;
    if (i >= NN) return;
    const float4* qr = (const float4*)(q + (size_t)i * 8);
    float4 v0 = qr[0], v1 = qr[1];
    float v[8] = {v0.x, v0.y, v0.z, v0.w, v1.x, v1.y, v1.z, v1.w};
    float m = v[0];
#pragma unroll
    for (int c = 1; c < 8; ++c) m = fmaxf(m, v[c]);
    float s = 0.0f;
#pragma unroll
    for (int c = 0; c < 8; ++c) { v[c] = __expf(v[c] - m); s += v[c]; }
    float inv = 1.0f / s;
#pragma unroll
    for (int c = 0; c < 8; ++c) smT[(size_t)c * NN + i] = (__bf16)(v[c] * inv);
}

// -------------------------------------------------------------------------
// 5) message = Kc @ sm via v_wmma_f32_16x16x32_bf16; fused compatibility +
//    q update. One block (4 waves) per 16-row tile; waves split the K dim.
// -------------------------------------------------------------------------
__global__ __launch_bounds__(128) void crf_wmma_kernel(const __bf16* __restrict__ Kc,
                                                       const __bf16* __restrict__ smT,
                                                       const float* __restrict__ uf,
                                                       const float* __restrict__ comp,
                                                       float* __restrict__ qout) {
    __shared__ float partial[4][16][16];
    __shared__ float msg[16][16];
    const int tid  = threadIdx.x;
    const int wave = tid >> 5;
    const int lane = tid & 31;
    const int half = lane >> 4;     // lane group (0: lanes 0-15, 1: lanes 16-31)
    const int mrow = lane & 15;     // A: row M ; B: column n
    const int i0 = blockIdx.x * 16;

    // A tile base: Kc[i0+M][ j0 + half*16 .. ] -- 16 contiguous K values per lane
    const __bf16* Arow = Kc  + (size_t)(i0 + mrow) * NN + half * 16;
    // B tile base: smT[n][ j0 + half*16 .. ] -- 16 contiguous K values per lane
    const __bf16* Brow = smT + (size_t)mrow * NN + half * 16;

    union U { uint4 u[2]; v16bf v; };
    v8f acc0 = {};
    v8f acc1 = {};
    const int jBeg = wave * (NN / 4);   // 2304 columns per wave
    const int jEnd = jBeg + (NN / 4);
    for (int j0 = jBeg; j0 < jEnd; j0 += 64) {
        __builtin_prefetch((const void*)(Arow + j0 + 512), 0, 1);
        U a, b;
        a.u[0] = *(const uint4*)(Arow + j0);
        a.u[1] = *(const uint4*)(Arow + j0 + 8);
        b.u[0] = *(const uint4*)(Brow + j0);
        b.u[1] = *(const uint4*)(Brow + j0 + 8);
        acc0 = __builtin_amdgcn_wmma_f32_16x16x32_bf16(
            false, a.v, false, b.v, (short)0, acc0, false, false);
        U a2, b2;
        a2.u[0] = *(const uint4*)(Arow + j0 + 32);
        a2.u[1] = *(const uint4*)(Arow + j0 + 40);
        b2.u[0] = *(const uint4*)(Brow + j0 + 32);
        b2.u[1] = *(const uint4*)(Brow + j0 + 40);
        acc1 = __builtin_amdgcn_wmma_f32_16x16x32_bf16(
            false, a2.v, false, b2.v, (short)0, acc1, false, false);
    }
    v8f acc = acc0 + acc1;

    // D layout: lane L, VGPR v -> msg[M = v + 8*(L>>4)][n = L&15]
#pragma unroll
    for (int v = 0; v < 8; ++v)
        partial[wave][v + 8 * half][mrow] = acc[v];
    __syncthreads();

    for (int e = tid; e < 256; e += 128) {
        int M = e >> 4, n = e & 15;
        msg[M][n] = partial[0][M][n] + partial[1][M][n] +
                    partial[2][M][n] + partial[3][M][n];
    }
    __syncthreads();

    // Fused compatibility transform + q update (channels 0..7 only)
    {
        int M = tid >> 3, c = tid & 7;
        float m = msg[M][c];
        float S = 0.0f;
#pragma unroll
        for (int cc = 0; cc < 8; ++cc) S += msg[M][cc];
        float c0 = comp[0], c1 = comp[1];
        int row = i0 + M;
        qout[(size_t)row * 8 + c] =
            uf[(size_t)row * 8 + c] - (c0 * m + c1 * (S - m));
    }
}

// -------------------------------------------------------------------------
// 6) Final transpose [N,C] -> [C,H,W]
// -------------------------------------------------------------------------
__global__ void crf_writeback_kernel(const float* __restrict__ q,
                                     float* __restrict__ out) {
    int idx = blockIdx.x * blockDim.x + threadIdx.x;
    if (idx >= NN * CC) return;
    int c = idx / NN, i = idx - c * NN;
    out[idx] = q[(size_t)i * 8 + c];
}

extern "C" void kernel_launch(void* const* d_in, const int* in_sizes, int n_in,
                              void* d_out, int out_size, void* d_ws, size_t ws_size,
                              hipStream_t stream) {
    const float* unary = (const float*)d_in[0];
    const float* rgb   = (const float*)d_in[1];
    const float* wsp   = (const float*)d_in[2];
    const float* wbi   = (const float*)d_in[3];
    const float* comp  = (const float*)d_in[4];

    char* ws = (char*)d_ws;
    __bf16* Kc  = (__bf16*)(ws + OFF_KC);
    float*  fb  = (float*) (ws + OFF_FB);
    float*  uf  = (float*) (ws + OFF_UF);
    float*  q   = (float*) (ws + OFF_Q);
    __bf16* smT = (__bf16*)(ws + OFF_SMT);
    float*  ns  = (float*) (ws + OFF_NS);
    float*  nb  = (float*) (ws + OFF_NB);

    crf_prep_kernel<<<(NN + 255) / 256, 256, 0, stream>>>(unary, rgb, fb, uf, q, smT);
    crf_norm_kernel<<<NN, 256, 0, stream>>>(fb, ns, nb);
    crf_kc_kernel<<<NN, 256, 0, stream>>>(fb, ns, nb, wsp, wbi, Kc);

    for (int it = 0; it < NITER; ++it) {
        crf_softmax_kernel<<<(NN + 255) / 256, 256, 0, stream>>>(q, smT);
        crf_wmma_kernel<<<NN / 16, 128, 0, stream>>>(Kc, smT, uf, comp, q);
    }

    crf_writeback_kernel<<<(NN * CC + 255) / 256, 256, 0, stream>>>(q, (float*)d_out);
}
